// MoELayer_20976620274021
// MI455X (gfx1250) — matrix operand order repaired and checked by reading, not compiled, not run on hardware
//
#include <hip/hip_runtime.h>
#include <math.h>

// ---------------- problem constants ----------------
#define E_ 8
#define D_ 1024
#define H_ 4096
#define T_ 4096          // B*S tokens
#define MTILE 32         // tokens per block tile
#define TILES_PER_E (T_ / MTILE)   // 128 (worst case: all tokens on one expert)
#define HCHUNK 128       // hidden chunk staged through LDS
#define XPAD 8           // bf16 row pad: (D_+8)*2 = 2064B; 2064 % 256 = 16 -> banks spread

#define W1_KT (D_ / 32)            // 32  k-tiles per expert (GEMM1)
#define W1_NT (H_ / 16)            // 256 n-tiles per expert
#define W1_TPE (W1_KT * W1_NT)     // 8192 tiles/expert
#define W2_KT (H_ / 32)            // 128
#define W2_NT (D_ / 16)            // 64
#define W2_TPE (W2_KT * W2_NT)     // 8192

typedef __attribute__((ext_vector_type(16))) __bf16        v16bf;
typedef __attribute__((ext_vector_type(8)))  float         v8f;
typedef __attribute__((ext_vector_type(4)))  unsigned int  u32x4;

union V16U { v16bf v; u32x4 q[2]; };

// float -> bf16, round-to-nearest-even (pack/gate kernels and the FFN h-chunk store)
__device__ __forceinline__ __bf16 f2bf(float f) {
    unsigned u = __builtin_bit_cast(unsigned, f);
    unsigned r = (u + 0x7FFFu + ((u >> 16) & 1u)) >> 16;
    return __builtin_bit_cast(__bf16, (unsigned short)r);
}

// A-fragment (16x32 bf16, M x K), wave32 WMMA layout:
// lane L: row M = L%16; elems 0..7 = K[k0+lh*8 ..], elems 8..15 = K[k0+16+lh*8 ..]
__device__ __forceinline__ v16bf load_a16(const __bf16* row, int k0, int lh) {
    V16U r;
    r.q[0] = *(const u32x4*)(row + k0 + lh * 8);
    r.q[1] = *(const u32x4*)(row + k0 + 16 + lh * 8);
    return r.v;
}

// B-fragment from a pre-packed 1KB tile: lane l's 16 bf16 at tile + l*16 elems.
// Whole wave reads one contiguous 1KB block -> two coalesced b128 loads per lane.
__device__ __forceinline__ v16bf load_b_packed(const __bf16* tile, int lane) {
    V16U r;
    const u32x4* p = (const u32x4*)(tile + lane * 16);
    r.q[0] = p[0];
    r.q[1] = p[1];
    return r.v;
}

__device__ __forceinline__ v8f wmma_bf16(v16bf a, v16bf b, v8f c) {
    return __builtin_amdgcn_wmma_f32_16x16x32_bf16(false, a, false, b, (short)0, c,
                                                   false, false);
}

// ---------------- kernel 0: reset per-expert counters ----------------
__global__ void zero_cnt_kernel(int* __restrict__ cnt) {
    if (threadIdx.x < E_) cnt[threadIdx.x] = 0;
}

// ---------------- kernel P: pack fp32 [E][K][N] weights into bf16 WMMA B-tiles -------
// One wave per 32x16 tile; lane l's fragment slice (col n = l%16, K half = l/16)
// stored contiguously at tile_base + l*32 bytes.
// nt_major=1: per-expert tile order [nt][kt] (W1, GEMM1 streams kt)
// nt_major=0: per-expert tile order [kt][nt] (W2, GEMM2 streams nt)
__global__ void __launch_bounds__(256) pack_b_kernel(
        const float* __restrict__ src, __bf16* __restrict__ dst,
        int K, int N, int nt_major) {
    const int tile = blockIdx.x * 8 + (int)(threadIdx.x >> 5);
    const int lane = threadIdx.x & 31;
    const int KT = K >> 5, NT = N >> 4;
    const int tpe = KT * NT;
    const int e = tile / tpe;
    const int r = tile % tpe;               // dst tile index within expert
    int kt, nt;
    if (nt_major) { kt = r % KT; nt = r / KT; }
    else          { nt = r % NT; kt = r / NT; }

    const float* s = src + ((size_t)e * K + (size_t)(kt * 32 + (lane >> 4) * 8)) * N
                         + nt * 16 + (lane & 15);
    V16U u;
#pragma unroll
    for (int i = 0; i < 16; ++i) u.v[i] = f2bf(s[(size_t)(i < 8 ? i : i + 8) * N]);  // coalesced across lanes

    __bf16* d = dst + ((size_t)e * tpe + r) * 512 + lane * 16;
    *(u32x4*)(d)     = u.q[0];
    *(u32x4*)(d + 8) = u.q[1];
}

// ---------------- kernel 1: gate + routing + x->bf16 ----------------
__global__ void __launch_bounds__(256) gate_kernel(
        const float* __restrict__ x, const float* __restrict__ Wg,
        const float* __restrict__ bg,
        __bf16* __restrict__ xbf, float* __restrict__ gw,
        int* __restrict__ perm, int* __restrict__ cnt,
        int* __restrict__ idx_out) {
    const int wave = threadIdx.x >> 5;
    const int lane = threadIdx.x & 31;
    const int t = blockIdx.x * 8 + wave;

    const float* xr  = x   + (size_t)t * D_;
    __bf16*      xbr = xbf + (size_t)t * D_;

    float acc[E_];
#pragma unroll
    for (int e = 0; e < E_; ++e) acc[e] = 0.0f;

    for (int d = lane; d < D_; d += 32) {
        float xv = xr[d];
        xbr[d] = f2bf(xv);
        const float* wr = Wg + (size_t)d * E_;
#pragma unroll
        for (int e = 0; e < E_; ++e) acc[e] += xv * wr[e];
    }

#pragma unroll
    for (int e = 0; e < E_; ++e) {
        float v = acc[e];
#pragma unroll
        for (int off = 16; off > 0; off >>= 1) v += __shfl_xor(v, off, 32);
        acc[e] = v;
    }

    if (lane == 0) {
        float lg[E_];
#pragma unroll
        for (int e = 0; e < E_; ++e) lg[e] = acc[e] + bg[e];
        int i0 = 0;
#pragma unroll
        for (int e = 1; e < E_; ++e) if (lg[e] > lg[i0]) i0 = e;
        int i1 = (i0 == 0) ? 1 : 0;
#pragma unroll
        for (int e = 0; e < E_; ++e) if (e != i0 && lg[e] > lg[i1]) i1 = e;
        float mx = lg[i0];
        float e0 = __expf(lg[i0] - mx), e1 = __expf(lg[i1] - mx);
        float inv = 1.0f / (e0 + e1);
        gw[t * 2 + 0] = e0 * inv;
        gw[t * 2 + 1] = e1 * inv;
        idx_out[t * 2 + 0] = i0;
        idx_out[t * 2 + 1] = i1;
        int p0 = atomicAdd(&cnt[i0], 1);
        perm[(size_t)i0 * T_ + p0] = (t << 1);
        int p1 = atomicAdd(&cnt[i1], 1);
        perm[(size_t)i1 * T_ + p1] = (t << 1) | 1;
    }
}

// ---------------- kernel 2: grouped fused FFN with packed bf16 weights ----------------
// x tile gathered to LDS once per block; A-fragments come from LDS (ds_load_b128),
// B-fragments stream from packed global tiles (global_load_b128), so the VMEM and DS
// pipes split the load traffic around the WMMAs.
__global__ void __launch_bounds__(256) moe_ffn_kernel(
        const __bf16* __restrict__ xbf,
        const __bf16* __restrict__ W1p, const float* __restrict__ b1,
        const __bf16* __restrict__ W2p, const float* __restrict__ b2,
        const int* __restrict__ perm, const int* __restrict__ cnt,
        float* __restrict__ ybuf) {
    __shared__ __attribute__((aligned(16))) __bf16 sx[MTILE][D_ + XPAD];     // 66 KB
    __shared__ __attribute__((aligned(16))) __bf16 sh[MTILE][HCHUNK + XPAD]; // 8.7 KB
    __shared__ int spair[MTILE];

    const int e     = blockIdx.x / TILES_PER_E;
    const int tile  = blockIdx.x % TILES_PER_E;
    const int cnt_e = cnt[e];
    const int m0    = tile * MTILE;
    if (m0 >= cnt_e) return;
    const int rem  = cnt_e - m0;
    const int rows = (rem < MTILE) ? rem : MTILE;

    if (threadIdx.x < MTILE) {
        int m = m0 + (int)threadIdx.x;
        spair[threadIdx.x] = (m < cnt_e) ? perm[(size_t)e * T_ + m] : 0;
    }
    __syncthreads();

    // gather 32 token rows (bf16) into LDS; padded rows alias row 0 (finite garbage,
    // results discarded; control flow stays convergent so EXEC is all-ones at WMMA)
    {
        int r   = (int)threadIdx.x >> 3;         // 0..31 row
        int seg = ((int)threadIdx.x & 7) * 128;  // column base, 128 bf16 per thread
        int msel = (r < rows) ? r : 0;
        const __bf16* srcw = xbf + (size_t)(spair[msel] >> 1) * D_ + seg;
        __bf16* dstw = &sx[r][seg];
#pragma unroll
        for (int i = 0; i < 8; ++i)
            *(u32x4*)(dstw + i * 16) = *(const u32x4*)(srcw + i * 16);
    }
    __syncthreads();

    const int w    = threadIdx.x >> 5;   // wave 0..7 -> owns D cols [w*128, w*128+128)
    const int lane = threadIdx.x & 31;
    const int lh   = lane >> 4;
    const int nl   = lane & 15;
    const int hi8  = lh ? 8 : 0;

    const __bf16* W1e = W1p + (size_t)e * W1_TPE * 512;
    const __bf16* W2e = W2p + (size_t)e * W2_TPE * 512;

    v8f acc[2][8];
#pragma unroll
    for (int mb = 0; mb < 2; ++mb)
#pragma unroll
        for (int nt = 0; nt < 8; ++nt) { v8f z = {}; acc[mb][nt] = z; }

    for (int hc = 0; hc < H_ / HCHUNK; ++hc) {
        const int hbase = hc * HCHUNK;
        const int hcol  = hbase + w * 16 + nl;

        // ---- GEMM1: h[32x128] = relu(x * W1[:,chunk] + b1); this wave: 16 cols ----
        const __bf16* bt1 = W1e + (size_t)(hc * 8 + w) * W1_KT * 512;  // 32KB kt stream
        if (hc + 1 < H_ / HCHUNK) {   // prefetch next chunk's tile streams
            __builtin_prefetch(bt1 + (size_t)8 * W1_KT * 512 + lane * 16, 0, 1);
            __builtin_prefetch(W2e + ((size_t)((hbase >> 5) + 4) * W2_NT + w * 8) * 512
                                   + lane * 16, 0, 1);
        }
        v8f c0 = {}, c1 = {};
#pragma unroll 4
        for (int kt = 0; kt < W1_KT; ++kt) {
            v16bf a0 = load_a16(&sx[nl][0],      kt * 32, lh);  // ds_load_b128 x2
            v16bf a1 = load_a16(&sx[16 + nl][0], kt * 32, lh);
            v16bf b  = load_b_packed(bt1 + (size_t)kt * 512, lane);
            c0 = wmma_bf16(a0, b, c0);
            c1 = wmma_bf16(a1, b, c1);
        }
        float b1v = b1[(size_t)e * H_ + hcol];
#pragma unroll
        for (int j = 0; j < 8; ++j) {
            float v0 = c0[j] + b1v; v0 = (v0 > 0.0f) ? v0 : 0.0f;
            float v1 = c1[j] + b1v; v1 = (v1 > 0.0f) ? v1 : 0.0f;
            sh[j + hi8][w * 16 + nl]      = f2bf(v0);   // C layout: vgpr j -> M = j + hi8
            sh[16 + j + hi8][w * 16 + nl] = f2bf(v1);
        }
        __syncthreads();   // h chunk visible to all waves

        // ---- GEMM2: acc += h[32x128] * W2[chunk, w*128 : w*128+128) ----
#pragma unroll
        for (int kq = 0; kq < 4; ++kq) {
            v16bf a0 = load_a16(&sh[nl][0],      kq * 32, lh);
            v16bf a1 = load_a16(&sh[16 + nl][0], kq * 32, lh);
            const __bf16* bt2 = W2e + ((size_t)((hbase >> 5) + kq) * W2_NT + w * 8) * 512;
#pragma unroll
            for (int nt = 0; nt < 8; ++nt) {
                v16bf b = load_b_packed(bt2 + (size_t)nt * 512, lane);
                acc[0][nt] = wmma_bf16(a0, b, acc[0][nt]);
                acc[1][nt] = wmma_bf16(a1, b, acc[1][nt]);
            }
        }
        __syncthreads();   // all waves done reading sh before next chunk overwrites
    }

    // ---- epilogue: y = acc + b2, scattered per (token, k) pair ----
#pragma unroll
    for (int mb = 0; mb < 2; ++mb) {
#pragma unroll
        for (int j = 0; j < 8; ++j) {
            int m = mb * 16 + j + hi8;
            if (m < rows) {
                int pair = spair[m];
                float* yr = ybuf + (size_t)pair * D_ + w * 128;
#pragma unroll
                for (int nt = 0; nt < 8; ++nt) {
                    float b2v = b2[(size_t)e * D_ + w * 128 + nt * 16 + nl];
                    yr[nt * 16 + nl] = acc[mb][nt][j] + b2v;
                }
            }
        }
    }
}

// ---------------- kernel 3: out[t] = g0*y(t,0) + g1*y(t,1) ----------------
__global__ void __launch_bounds__(256) combine_kernel(
        const float* __restrict__ ybuf, const float* __restrict__ gw,
        float* __restrict__ out) {
    int i = (blockIdx.x * 256 + (int)threadIdx.x) * 4;
    int t = i >> 10;
    int d = i & (D_ - 1);
    float g0 = gw[t * 2 + 0];
    float g1 = gw[t * 2 + 1];
    const float* y0 = ybuf + ((size_t)t * 2) * D_ + d;
    const float* y1 = y0 + D_;
    float4 a = *(const float4*)y0;
    float4 b = *(const float4*)y1;
    float4 o;
    o.x = g0 * a.x + g1 * b.x;
    o.y = g0 * a.y + g1 * b.y;
    o.z = g0 * a.z + g1 * b.z;
    o.w = g0 * a.w + g1 * b.w;
    *(float4*)(out + (size_t)t * D_ + d) = o;
}

// ---------------- launch ----------------
extern "C" void kernel_launch(void* const* d_in, const int* in_sizes, int n_in,
                              void* d_out, int out_size, void* d_ws, size_t ws_size,
                              hipStream_t stream) {
    const float* x  = (const float*)d_in[0];
    const float* Wg = (const float*)d_in[1];
    const float* bg = (const float*)d_in[2];
    const float* W1 = (const float*)d_in[3];
    const float* b1 = (const float*)d_in[4];
    const float* W2 = (const float*)d_in[5];
    const float* b2 = (const float*)d_in[6];

    float* out     = (float*)d_out;
    int*   idx_out = (int*)d_out + (size_t)T_ * D_;   // tuple output #2 (int32 region)

    char* ws = (char*)d_ws;
    size_t off = 0;
    auto take = [&](size_t bytes) {
        char* p = ws + off;
        off += (bytes + 255) & ~(size_t)255;
        return p;
    };
    __bf16* xbf  = (__bf16*)take((size_t)T_ * D_ * 2);            // 8 MB
    float*  gw   = (float*) take((size_t)T_ * 2 * 4);             // 32 KB
    int*    perm = (int*)   take((size_t)E_ * T_ * 4);            // 128 KB
    int*    cnt  = (int*)   take((size_t)E_ * 4);
    float*  ybuf = (float*) take((size_t)T_ * 2 * D_ * 4);        // 32 MB
    __bf16* W1p  = (__bf16*)take((size_t)E_ * D_ * H_ * 2);       // 67 MB packed
    __bf16* W2p  = (__bf16*)take((size_t)E_ * H_ * D_ * 2);       // 67 MB packed

    zero_cnt_kernel<<<1, 32, 0, stream>>>(cnt);
    pack_b_kernel<<<E_ * W1_TPE / 8, 256, 0, stream>>>(W1, W1p, D_, H_, 1);
    pack_b_kernel<<<E_ * W2_TPE / 8, 256, 0, stream>>>(W2, W2p, H_, D_, 0);
    gate_kernel<<<T_ / 8, 256, 0, stream>>>(x, Wg, bg, xbf, gw, perm, cnt, idx_out);
    moe_ffn_kernel<<<E_ * TILES_PER_E, 256, 0, stream>>>(xbf, W1p, b1, W2p, b2, perm, cnt, ybuf);
    combine_kernel<<<(T_ * D_) / (256 * 4), 256, 0, stream>>>(ybuf, gw, out);

    (void)in_sizes; (void)n_in; (void)out_size; (void)ws_size;
}
